// AspectOrientedDepGCN_30365418783493
// MI455X (gfx1250) — compile-verified, matrix operand order checked by reading, not run on hardware
//
#include <hip/hip_runtime.h>
#include <hip/hip_bf16.h>

#define N_NODES 50000
#define N_EDGES 100000
#define DIM 768
#define LN_EPS 1e-5f

typedef float v8f  __attribute__((ext_vector_type(8)));
typedef __bf16 v8bf  __attribute__((ext_vector_type(8)));
typedef __bf16 v16bf __attribute__((ext_vector_type(16)));

// ---------------------------------------------------------------- helpers
__device__ __forceinline__ unsigned short f32_to_bf16_rne(float f) {
  unsigned u = __float_as_uint(f);
  unsigned r = u + 0x7FFFu + ((u >> 16) & 1u);
  return (unsigned short)(r >> 16);
}
__device__ __forceinline__ float bf16_bits_to_f32(unsigned short h) {
  return __uint_as_float(((unsigned)h) << 16);
}
__device__ __forceinline__ void split_bf16(float x, unsigned short& hi, unsigned short& lo) {
  hi = f32_to_bf16_rne(x);
  lo = f32_to_bf16_rne(x - bf16_bits_to_f32(hi));
}
__device__ __forceinline__ v8f wmma_bf16(v16bf a, v16bf b, v8f c) {
  return __builtin_amdgcn_wmma_f32_16x16x32_bf16(false, a, false, b, (short)0, c,
                                                 false, false);
}
// Load one 16x16x32 bf16 A/B fragment for this lane from LDS.
// Layout: [idx][k] bf16 with row stride `strd`; lane<16 holds K 0..7 & 16..23,
// lane>=16 holds K 8..15 & 24..31 (two aligned b128 loads).
__device__ __forceinline__ v16bf ld_frag(const unsigned short* base, int strd,
                                         int idx, int lh) {
  const v8bf c0 = *(const v8bf*)(base + idx * strd + lh * 8);
  const v8bf c1 = *(const v8bf*)(base + idx * strd + 16 + lh * 8);
  return __builtin_shufflevector(c0, c1, 0, 1, 2, 3, 4, 5, 6, 7, 8, 9, 10, 11,
                                 12, 13, 14, 15);
}

// ---------------------------------------------------------------- utilities
__global__ __launch_bounds__(256) void zero_kernel(float* __restrict__ p, long n4) {
  long i = (long)blockIdx.x * blockDim.x + threadIdx.x;
  long stride = (long)gridDim.x * blockDim.x;
  float4* p4 = (float4*)p;
  for (; i < n4; i += stride) p4[i] = make_float4(0.f, 0.f, 0.f, 0.f);
}

// Wt[n][k] (split bf16) = W[k][n]  — run once per weight matrix (L2-resident)
__global__ __launch_bounds__(256) void convert_wt_kernel(
    const float* __restrict__ W, unsigned short* __restrict__ Hi,
    unsigned short* __restrict__ Lo) {
  int idx = blockIdx.x * 256 + threadIdx.x;   // idx = n*768 + k
  int n = idx / DIM, k = idx % DIM;
  unsigned short h, l;
  split_bf16(W[(long)k * DIM + n], h, l);
  Hi[idx] = h;
  Lo[idx] = l;
}

// arow[n] = gate_b[n] + sum_k aspect[k] * gate_w[D + k][n]   (constant)
__global__ __launch_bounds__(256) void aspect_row_kernel(
    const float* __restrict__ aspect, const float* __restrict__ gate_w,
    const float* __restrict__ gate_b, float* __restrict__ arow) {
  int n = blockIdx.x * blockDim.x + threadIdx.x;
  if (n >= DIM) return;
  float acc = gate_b[n];
  for (int k = 0; k < DIM; ++k)
    acc += aspect[k] * gate_w[(long)(DIM + k) * DIM + n];
  arow[n] = acc;
}

// agg[dst] += x[src] over edges; float4 gather, native global fp32 atomics
__global__ __launch_bounds__(256) void scatter_add_kernel(
    const int* __restrict__ ei, const float* __restrict__ xsrc,
    float* __restrict__ agg) {
  const int D4 = DIM / 4;
  long tid = (long)blockIdx.x * blockDim.x + threadIdx.x;
  if (tid >= (long)N_EDGES * D4) return;
  int e = (int)(tid / D4);
  int c = (int)(tid % D4) * 4;
  int s = ei[e];
  int d = ei[N_EDGES + e];
  const float4 v = *(const float4*)(xsrc + (long)s * DIM + c);
  float* dst = agg + (long)d * DIM + c;
  unsafeAtomicAdd(dst + 0, v.x);
  unsafeAtomicAdd(dst + 1, v.y);
  unsafeAtomicAdd(dst + 2, v.z);
  unsafeAtomicAdd(dst + 3, v.w);
}

// ----------------- GEMM1: xgcn = relu(agg @ W + b), emitted as split-bf16
// A fp32 [N,D] converted to split-bf16 in LDS; B pre-split transposed bf16.
#define G1_BM 128
#define G1_BN 64
#define BK 32
#define ARS1 40  // bf16 k-stride, mult of 8; 20*l mod 64 distinct -> no conflicts
#define BRS 40

__global__ __launch_bounds__(256) void gemm1_relu_kernel(
    const float* __restrict__ A, const unsigned short* __restrict__ WtHi,
    const unsigned short* __restrict__ WtLo, const float* __restrict__ bias,
    unsigned short* __restrict__ XgHi, unsigned short* __restrict__ XgLo) {
  __shared__ unsigned short Ah[G1_BM * ARS1], Al[G1_BM * ARS1];
  __shared__ unsigned short Bh[G1_BN * BRS], Bl[G1_BN * BRS];
  const int t = threadIdx.x;
  const int lane = t & 31, wv = t >> 5;
  const int lh = lane >> 4, ll = lane & 15;
  const int m0 = blockIdx.y * G1_BM;
  const int n0 = blockIdx.x * G1_BN;

  v8f acc[4] = {};

  for (int k0 = 0; k0 < DIM; k0 += BK) {
    // A tile: 128x32 f32 -> split bf16 (1024 float4, 4 per thread)
    #pragma unroll
    for (int it = 0; it < 4; ++it) {
      int idx = t + 256 * it;
      int row = idx >> 3, c4 = (idx & 7) * 4;
      int grow = m0 + row;
      if (grow >= N_NODES) grow = N_NODES - 1;
      const float4 v = *(const float4*)(A + (long)grow * DIM + k0 + c4);
      ushort4 h, l;
      split_bf16(v.x, h.x, l.x);
      split_bf16(v.y, h.y, l.y);
      split_bf16(v.z, h.z, l.z);
      split_bf16(v.w, h.w, l.w);
      *(ushort4*)&Ah[row * ARS1 + c4] = h;
      *(ushort4*)&Al[row * ARS1 + c4] = l;
    }
    // B tile: 64 cols x 32 k bf16, hi+lo (uint4 copies, 1 each per thread)
    {
      int col = t >> 2, ch = (t & 3) * 8;
      *(uint4*)&Bh[col * BRS + ch] =
          *(const uint4*)(WtHi + (long)(n0 + col) * DIM + k0 + ch);
      *(uint4*)&Bl[col * BRS + ch] =
          *(const uint4*)(WtLo + (long)(n0 + col) * DIM + k0 + ch);
      if (k0 + BK < DIM)
        __builtin_prefetch(WtHi + (long)(n0 + col) * DIM + k0 + BK + ch, 0, 0);
    }
    __syncthreads();
    const int row = 16 * wv + ll;
    v16bf ahi = ld_frag(Ah, ARS1, row, lh);
    v16bf alo = ld_frag(Al, ARS1, row, lh);
    #pragma unroll
    for (int ct = 0; ct < 4; ++ct) {
      int col = ct * 16 + ll;
      v16bf bhi = ld_frag(Bh, BRS, col, lh);
      v16bf blo = ld_frag(Bl, BRS, col, lh);
      acc[ct] = wmma_bf16(ahi, bhi, acc[ct]);
      acc[ct] = wmma_bf16(ahi, blo, acc[ct]);
      acc[ct] = wmma_bf16(alo, bhi, acc[ct]);
    }
    __syncthreads();
  }

  #pragma unroll
  for (int ct = 0; ct < 4; ++ct) {
    int col = n0 + ct * 16 + ll;
    float bv = bias[col];
    #pragma unroll
    for (int v = 0; v < 8; ++v) {
      int row = m0 + 16 * wv + v + 8 * lh;
      if (row < N_NODES) {
        float val = acc[ct][v] + bv;
        val = val > 0.f ? val : 0.f;
        unsigned short h, l;
        split_bf16(val, h, l);
        XgHi[(long)row * DIM + col] = h;
        XgLo[(long)row * DIM + col] = l;
      }
    }
  }
}

// -------- GEMM2 (gate) + sigmoid + gated mix + LayerNorm; full rows per block.
// Two passes over B (hi then lo) keep static LDS at ~64.7KB (4 WG/WGP).
#define G2_BM 32
#define ARS2 24  // mult of 8; 12*l mod 64 distinct -> conflict-free reads

__global__ __launch_bounds__(256) void gemm2_epilogue_kernel(
    const unsigned short* __restrict__ XgHi, const unsigned short* __restrict__ XgLo,
    const unsigned short* __restrict__ GtHi, const unsigned short* __restrict__ GtLo,
    const float* __restrict__ arow, const float* __restrict__ Xold,
    const float* __restrict__ gamma, const float* __restrict__ beta,
    float* __restrict__ outp) {
  __shared__ unsigned short Ah[G2_BM * ARS2], Al[G2_BM * ARS2];
  __shared__ unsigned short Bs[DIM * BRS / 8 * 8];  // 768*40 bf16 = 61440B
  __shared__ float rowsum[G2_BM];
  __shared__ float rowsq[G2_BM];
  const int t = threadIdx.x;
  const int lane = t & 31, wv = t >> 5;
  const int rg = wv >> 2, cg = wv & 3;  // 2 row-groups x 4 col-groups
  const int lh = lane >> 4, ll = lane & 15;
  const int m0 = blockIdx.x * G2_BM;

  if (t < G2_BM) { rowsum[t] = 0.f; rowsq[t] = 0.f; }

  v8f acc[12] = {};

  #pragma unroll 1
  for (int pass = 0; pass < 2; ++pass) {
    const unsigned short* Bsrc = pass ? GtLo : GtHi;
    #pragma unroll 1
    for (int k0 = 0; k0 < DIM; k0 += BK) {
      // A tile: 32 rows x 32 k, split bf16 copies (uint4, 1 per thread)
      {
        int half = t >> 7;              // 0: hi, 1: lo
        int t2 = t & 127;
        int row = t2 >> 2, ch = (t2 & 3) * 8;
        int grow = m0 + row;
        if (grow >= N_NODES) grow = N_NODES - 1;
        const unsigned short* src = half ? XgLo : XgHi;
        unsigned short* dst = half ? Al : Ah;
        *(uint4*)&dst[row * ARS2 + ch] =
            *(const uint4*)(src + (long)grow * DIM + k0 + ch);
      }
      // B tile: 768 cols x 32 k bf16 from current pass (3072 uint4, 12 each)
      #pragma unroll
      for (int it = 0; it < 12; ++it) {
        int idx = t + 256 * it;
        int col = idx >> 2, ch = (idx & 3) * 8;
        *(uint4*)&Bs[col * BRS + ch] =
            *(const uint4*)(Bsrc + (long)col * DIM + k0 + ch);
        if (k0 + BK < DIM)
          __builtin_prefetch(Bsrc + (long)col * DIM + k0 + BK + ch, 0, 0);
      }
      __syncthreads();
      const int row = 16 * rg + ll;
      v16bf ahi = ld_frag(Ah, ARS2, row, lh);
      if (pass == 0) {
        v16bf alo = ld_frag(Al, ARS2, row, lh);
        #pragma unroll
        for (int ct = 0; ct < 12; ++ct) {
          v16bf bh = ld_frag(Bs, BRS, 192 * cg + ct * 16 + ll, lh);
          acc[ct] = wmma_bf16(ahi, bh, acc[ct]);
          acc[ct] = wmma_bf16(alo, bh, acc[ct]);
        }
      } else {
        #pragma unroll
        for (int ct = 0; ct < 12; ++ct) {
          v16bf bl = ld_frag(Bs, BRS, 192 * cg + ct * 16 + ll, lh);
          acc[ct] = wmma_bf16(ahi, bl, acc[ct]);
        }
      }
      __syncthreads();
    }
  }

  // gate = sigmoid(z + arow); y = g*xg + (1-g)*xold   (+ row stats)
  float ps[8] = {}, ps2[8] = {};
  #pragma unroll
  for (int ct = 0; ct < 12; ++ct) {
    int col = 192 * cg + ct * 16 + ll;
    float ar = arow[col];
    #pragma unroll
    for (int v = 0; v < 8; ++v) {
      int row = m0 + 16 * rg + v + 8 * lh;
      int crow = row < N_NODES ? row : N_NODES - 1;
      long off = (long)crow * DIM + col;
      float xg = bf16_bits_to_f32(XgHi[off]) + bf16_bits_to_f32(XgLo[off]);
      float xo = Xold[off];
      float g = 1.f / (1.f + __expf(-(acc[ct][v] + ar)));
      float y = g * xg + (1.f - g) * xo;
      acc[ct][v] = y;
      ps[v] += y;
      ps2[v] += y * y;
    }
  }
  #pragma unroll
  for (int v = 0; v < 8; ++v) {
    int r = 16 * rg + v + 8 * lh;
    atomicAdd(&rowsum[r], ps[v]);  // ds_add_f32
    atomicAdd(&rowsq[r], ps2[v]);
  }
  __syncthreads();

  const float invD = 1.f / (float)DIM;
  #pragma unroll
  for (int ct = 0; ct < 12; ++ct) {
    int col = 192 * cg + ct * 16 + ll;
    float gm = gamma[col], bt = beta[col];
    #pragma unroll
    for (int v = 0; v < 8; ++v) {
      int row = m0 + 16 * rg + v + 8 * lh;
      if (row < N_NODES) {
        int r = 16 * rg + v + 8 * lh;
        float mu = rowsum[r] * invD;
        float var = rowsq[r] * invD - mu * mu;
        outp[(long)row * DIM + col] =
            (acc[ct][v] - mu) * rsqrtf(var + LN_EPS) * gm + bt;
      }
    }
  }
}

// ---------------------------------------------------------------- launcher
extern "C" void kernel_launch(void* const* d_in, const int* in_sizes, int n_in,
                              void* d_out, int out_size, void* d_ws, size_t ws_size,
                              hipStream_t stream) {
  const float* tokens   = (const float*)d_in[0];
  const int*   ei       = (const int*)d_in[1];
  const float* aspect   = (const float*)d_in[2];
  const float* gcn_w    = (const float*)d_in[3];
  const float* gcn_b    = (const float*)d_in[4];
  const float* gate_w   = (const float*)d_in[5];
  const float* gate_b   = (const float*)d_in[6];
  const float* ln_gamma = (const float*)d_in[7];
  const float* ln_beta  = (const float*)d_in[8];
  float* out = (float*)d_out;

  const size_t ND = (size_t)N_NODES * DIM;
  const size_t WSZ = (size_t)DIM * DIM;
  float* agg = (float*)d_ws;                           // N*D f32
  unsigned short* xgh = (unsigned short*)(agg + ND);   // N*D bf16
  unsigned short* xgl = xgh + ND;                      // N*D bf16
  float* arow = (float*)(xgl + ND);                    // D f32
  unsigned short* wsplit = (unsigned short*)(arow + DIM);
  unsigned short* w1h = wsplit;                        // gcn layer0 Wt hi
  unsigned short* w1l = w1h + WSZ;
  unsigned short* w2h = w1l + WSZ;                     // gcn layer1 Wt hi
  unsigned short* w2l = w2h + WSZ;
  unsigned short* gth = w2l + WSZ;                     // gate top-block Wt hi
  unsigned short* gtl = gth + WSZ;

  const int cvtBlocks = (int)(WSZ / 256);
  convert_wt_kernel<<<cvtBlocks, 256, 0, stream>>>(gcn_w, w1h, w1l);
  convert_wt_kernel<<<cvtBlocks, 256, 0, stream>>>(gcn_w + WSZ, w2h, w2l);
  convert_wt_kernel<<<cvtBlocks, 256, 0, stream>>>(gate_w, gth, gtl);
  aspect_row_kernel<<<3, 256, 0, stream>>>(aspect, gate_w, gate_b, arow);

  const long n4 = (long)ND / 4;
  const long nscat = (long)N_EDGES * (DIM / 4);
  dim3 g1grid(DIM / G1_BN, (N_NODES + G1_BM - 1) / G1_BM);

  for (int layer = 0; layer < 2; ++layer) {
    const float* xcur = (layer == 0) ? tokens : out;
    zero_kernel<<<4096, 256, 0, stream>>>(agg, n4);
    scatter_add_kernel<<<(int)((nscat + 255) / 256), 256, 0, stream>>>(ei, xcur, agg);
    gemm1_relu_kernel<<<g1grid, 256, 0, stream>>>(
        agg, layer ? w2h : w1h, layer ? w2l : w1l, gcn_b + layer * DIM, xgh, xgl);
    gemm2_epilogue_kernel<<<(N_NODES + G2_BM - 1) / G2_BM, 256, 0, stream>>>(
        xgh, xgl, gth, gtl, arow, xcur, ln_gamma + layer * DIM,
        ln_beta + layer * DIM, out);
  }
}
